// DifferCrossAttn_11184094839168
// MI455X (gfx1250) — matrix operand order repaired and checked by reading, not compiled, not run on hardware
//
#include <hip/hip_runtime.h>

typedef __bf16 bf16;
typedef __attribute__((ext_vector_type(16))) __bf16 v16bf;
typedef __attribute__((ext_vector_type(8)))  __bf16 v8bf;
typedef __attribute__((ext_vector_type(8)))  float  v8f;
typedef __attribute__((ext_vector_type(4)))  float  v4f;
typedef __attribute__((ext_vector_type(4)))  int    v4i;

#define LDSROW 40        // 32 k halves + 8 pad per LDS tile row (80B = 16B-aligned stride)
#define LAMBDA_INIT 0.2f // 0.8 - 0.6*exp(-0.3*0)

// ---------------- CDNA5 async global->LDS (ASYNCcnt-tracked) ----------------
__device__ __forceinline__ void async_ld16(bf16* l, const bf16* g) {
#if __has_builtin(__builtin_amdgcn_global_load_async_to_lds_b128)
  __builtin_amdgcn_global_load_async_to_lds_b128(
      (__attribute__((address_space(1))) v4i*)g,
      (__attribute__((address_space(3))) v4i*)l, 0, 0);
#else
  unsigned off = (unsigned)(unsigned long long)l;  // low 32 bits = LDS offset
  asm volatile("global_load_async_to_lds_b128 %0, %1, off"
               :: "v"(off), "v"(g) : "memory");
#endif
}

#if __has_builtin(__builtin_amdgcn_s_wait_asynccnt)
#define WAIT_ASYNC(n) __builtin_amdgcn_s_wait_asynccnt(n)
#else
#define WAIT_ASYNC(n) asm volatile("s_wait_asynccnt %0" :: "i"(n) : "memory")
#endif

// ---------------- wave reductions (wave32) ----------------
__device__ __forceinline__ float wave_rmax(float v) {
  #pragma unroll
  for (int m = 16; m >= 1; m >>= 1) v = fmaxf(v, __shfl_xor(v, m, 32));
  return v;
}
__device__ __forceinline__ float wave_rsum(float v) {
  #pragma unroll
  for (int m = 16; m >= 1; m >>= 1) v += __shfl_xor(v, m, 32);
  return v;
}

// ---------------- WMMA helpers ----------------
__device__ __forceinline__ v8f wmma_bf16(v16bf a, v16bf b, v8f c) {
  return __builtin_amdgcn_wmma_f32_16x16x32_bf16(false, a, false, b, (short)0, c,
                                                 false, false);
}

// A fragment (16x32 bf16): lane L holds row m=L&15; halves 0..7 -> k=8*(L>>4)+0..7,
// halves 8..15 -> +16 (per ISA A-layout table).
__device__ __forceinline__ v16bf load_a_frag(const bf16* base, int ld) {
  const int lane = threadIdx.x & 31;
  const bf16* p = base + (lane & 15) * ld + ((lane >> 4) << 3);
  v8bf lo = *(const v8bf*)p;
  v8bf hi = *(const v8bf*)(p + 16);
  v16bf f;
  #pragma unroll
  for (int i = 0; i < 8; ++i) { f[i] = lo[i]; f[i + 8] = hi[i]; }
  return f;
}

// B fragment (32x16 bf16): lane L holds column n=L&15; halves 0..7 -> k=8*(L>>4)+0..7,
// halves 8..15 -> +16 (same k order as A).
__device__ __forceinline__ v16bf load_b_frag(const bf16* base, int ld) {
  const int lane = threadIdx.x & 31;
  const bf16* p = base + (lane & 15) * ld + ((lane >> 4) << 3);
  v8bf lo = *(const v8bf*)p;
  v8bf hi = *(const v8bf*)(p + 16);
  v16bf f;
  #pragma unroll
  for (int i = 0; i < 8; ++i) { f[i] = lo[i]; f[i + 8] = hi[i]; }
  return f;
}

// ---------------- lambda scalar ----------------
__global__ void __launch_bounds__(256)
lambda_kernel(const float* __restrict__ q1, const float* __restrict__ k1,
              const float* __restrict__ q2, const float* __restrict__ k2,
              float* __restrict__ lam) {
  __shared__ float r1[8], r2[8];
  const int t = threadIdx.x;                 // 256 == HEAD_DIM
  float s1 = wave_rsum(q1[t] * k1[t]);
  float s2 = wave_rsum(q2[t] * k2[t]);
  if ((t & 31) == 0) { r1[t >> 5] = s1; r2[t >> 5] = s2; }
  __syncthreads();
  if (t == 0) {
    float a = 0.f, b = 0.f;
    #pragma unroll
    for (int i = 0; i < 8; ++i) { a += r1[i]; b += r2[i]; }
    *lam = __expf(a) - __expf(b) + LAMBDA_INIT;
  }
}

// ---------------- f32 -> bf16 conversion ----------------
__global__ void __launch_bounds__(256)
cvt_kernel(const float* __restrict__ src, bf16* __restrict__ dst, int n4) {
  const int i = blockIdx.x * 256 + threadIdx.x;
  if (i < n4) {
    v4f v = *(const v4f*)(src + (size_t)i * 4);
    bf16* d = dst + (size_t)i * 4;
    #pragma unroll
    for (int j = 0; j < 4; ++j) d[j] = (bf16)v[j];
  }
}

// ---------------- NT GEMM: C[M,N] = scale * A[M,K] * B[N,K]^T ----------------
// 256 threads (8 waves), tile 128x64, K step 32, double-buffered LDS filled by
// async global->LDS DMA; each wave owns a 32x32 output (4 WMMAs per k-step).
// mode: 0 = bf16 row-major, 1 = f32 row-major, 2 = bf16 per-batch V-transpose.
__device__ __forceinline__ void stage_async(bf16* Asb, bf16* Bsb,
    const bf16* A, const bf16* B, int rowbase, int nbase, int K, int kb,
    int sr, int skg) {
  async_ld16(&Asb[sr * LDSROW + skg],        &A[(size_t)(rowbase + sr) * K + kb + skg]);
  async_ld16(&Asb[(sr + 64) * LDSROW + skg], &A[(size_t)(rowbase + sr + 64) * K + kb + skg]);
  async_ld16(&Bsb[sr * LDSROW + skg],        &B[(size_t)(nbase + sr) * K + kb + skg]);
}

__global__ void __launch_bounds__(256)
gemm_nt_bf16(const bf16* __restrict__ A, const bf16* __restrict__ B,
             bf16* __restrict__ Cb, float* __restrict__ Cf,
             int K, float scale, int mode) {
  __shared__ __align__(16) bf16 As[2][128 * LDSROW];
  __shared__ __align__(16) bf16 Bs[2][64 * LDSROW];
  const int t = threadIdx.x;
  const int wave = t >> 5, lane = t & 31;
  const int wm = wave >> 1, wn = wave & 1;          // 4x2 waves -> 128x64 tile
  const int rowbase = blockIdx.y * 128;
  const int nbase   = blockIdx.x * 64;
  const int N = gridDim.x * 64;
  const int sr = t >> 2, skg = (t & 3) << 3;        // staging: row, k-group (16B)
  const int nsteps = K >> 5;
  v8f acc00 = {}, acc01 = {}, acc10 = {}, acc11 = {};

  stage_async(As[0], Bs[0], A, B, rowbase, nbase, K, 0, sr, skg);
  stage_async(As[1], Bs[1], A, B, rowbase, nbase, K, 32, sr, skg);

  for (int s = 0; s < nsteps; ++s) {
    // async loads retire in order: waiting <=3 guarantees this step's 3 are done
    // while next step's 3 stay in flight.
    if (s < nsteps - 1) { WAIT_ASYNC(3); } else { WAIT_ASYNC(0); }
    __syncthreads();
    const bf16* Ab = As[s & 1];
    const bf16* Bb = Bs[s & 1];
    v16bf a0 = load_a_frag(&Ab[(wm * 32)      * LDSROW], LDSROW);
    v16bf a1 = load_a_frag(&Ab[(wm * 32 + 16) * LDSROW], LDSROW);
    v16bf b0 = load_b_frag(&Bb[(wn * 32)      * LDSROW], LDSROW);
    v16bf b1 = load_b_frag(&Bb[(wn * 32 + 16) * LDSROW], LDSROW);
    acc00 = wmma_bf16(a0, b0, acc00);
    acc01 = wmma_bf16(a0, b1, acc01);
    acc10 = wmma_bf16(a1, b0, acc10);
    acc11 = wmma_bf16(a1, b1, acc11);
    __syncthreads();
    if (s + 2 < nsteps)
      stage_async(As[s & 1], Bs[s & 1], A, B, rowbase, nbase, K, (s + 2) * 32, sr, skg);
  }

  const int cn = lane & 15, cm = (lane >> 4) << 3;
  #pragma unroll
  for (int r = 0; r < 8; ++r) {
    const int row0 = rowbase + wm * 32 + cm + r;
    const int col0 = nbase + wn * 32 + cn;
    const float v00 = acc00[r] * scale, v01 = acc01[r] * scale;
    const float v10 = acc10[r] * scale, v11 = acc11[r] * scale;
    if (mode == 0) {
      Cb[(size_t)row0 * N + col0]             = (bf16)v00;
      Cb[(size_t)row0 * N + col0 + 16]        = (bf16)v01;
      Cb[(size_t)(row0 + 16) * N + col0]      = (bf16)v10;
      Cb[(size_t)(row0 + 16) * N + col0 + 16] = (bf16)v11;
    } else if (mode == 1) {
      Cf[(size_t)row0 * N + col0]             = v00;
      Cf[(size_t)row0 * N + col0 + 16]        = v01;
      Cf[(size_t)(row0 + 16) * N + col0]      = v10;
      Cf[(size_t)(row0 + 16) * N + col0 + 16] = v11;
    } else {  // Vt[b][e][s] with b = row>>10, s = row&1023, e = col
      auto st = [&](int row, int col, float v) {
        Cb[((size_t)(row >> 10) * 512 + col) * 1024 + (row & 1023)] = (bf16)v;
      };
      st(row0, col0, v00); st(row0, col0 + 16, v01);
      st(row0 + 16, col0, v10); st(row0 + 16, col0 + 16, v11);
    }
  }
}

// ---------------- fused QK^T + softmax + differential attention ----------------
#define SROW 1032
__global__ void __launch_bounds__(256)
scores_softmax(const bf16* __restrict__ Qb, const bf16* __restrict__ Kb,
               const float* __restrict__ lamp, float* __restrict__ diff,
               bf16* __restrict__ Pb) {
  __shared__ __align__(16) float S[2 * 16 * SROW];   // 132 KB (< 320 KB WGP LDS)
  const int t = threadIdx.x;
  const int wave = t >> 5, lane = t & 31;
  const int grow = blockIdx.x * 16;                  // global query row base
  const int b = grow >> 12;                          // batch (NG = 4096)
  const size_t qoff = (size_t)grow * 512;
  const size_t koff = (size_t)b * 1024 * 512;

  for (int h = 0; h < 2; ++h) {
    for (int i = 0; i < 8; ++i) {
      const int stile = wave * 8 + i;
      v8f acc = {};
      #pragma unroll
      for (int ks = 0; ks < 8; ++ks) {
        const int kcol = h * 256 + ks * 32;
        v16bf a  = load_a_frag(Qb + qoff + kcol, 512);
        v16bf bb = load_b_frag(Kb + koff + (size_t)stile * 16 * 512 + kcol, 512);
        acc = wmma_bf16(a, bb, acc);
      }
      float* sp = &S[(h * 16) * SROW + stile * 16];
      const int cn = lane & 15, cm = (lane >> 4) << 3;
      #pragma unroll
      for (int r = 0; r < 8; ++r) sp[(cm + r) * SROW + cn] = acc[r];
    }
  }
  __syncthreads();

  #pragma unroll
  for (int rr = 0; rr < 4; ++rr) {
    float* sp = &S[(wave * 4 + rr) * SROW];
    float mx = -3.0e38f;
    #pragma unroll
    for (int j = 0; j < 32; ++j) mx = fmaxf(mx, sp[lane + j * 32]);
    mx = wave_rmax(mx);
    float sum = 0.f;
    #pragma unroll
    for (int j = 0; j < 32; ++j) {
      float e = __expf(sp[lane + j * 32] - mx);
      sp[lane + j * 32] = e;
      sum += e;
    }
    const float inv = 1.0f / wave_rsum(sum);
    #pragma unroll
    for (int j = 0; j < 32; ++j) sp[lane + j * 32] *= inv;
  }
  __syncthreads();

  const float lam = *lamp;
  for (int idx = t; idx < 16 * 1024; idx += 256) {
    const int q = idx >> 10, s = idx & 1023;
    const float d = S[q * SROW + s] - lam * S[(16 + q) * SROW + s];
    const size_t o = (size_t)(grow + q) * 1024 + s;
    diff[o] = d;           // f32 output
    Pb[o]   = (bf16)d;     // bf16 feed for P@V GEMM
  }
}

// ---------------- RMSNorm * weight * (1 - lambda_init) -> bf16 ----------------
__global__ void __launch_bounds__(256)
rms_kernel(const float* __restrict__ Ypre, const float* __restrict__ w,
           bf16* __restrict__ Yb) {
  __shared__ float red[8];
  const int row = blockIdx.x, t = threadIdx.x;
  const float* x = Ypre + (size_t)row * 512;
  const float v0 = x[t], v1 = x[t + 256];
  float ss = wave_rsum(v0 * v0 + v1 * v1);
  if ((t & 31) == 0) red[t >> 5] = ss;
  __syncthreads();
  float tot = 0.f;
  #pragma unroll
  for (int i = 0; i < 8; ++i) tot += red[i];
  const float r = rsqrtf(tot * (1.0f / 512.0f) + 1e-5f) * (1.0f - LAMBDA_INIT);
  Yb[(size_t)row * 512 + t]       = (bf16)(v0 * r * w[t]);
  Yb[(size_t)row * 512 + t + 256] = (bf16)(v1 * r * w[t + 256]);
}

// ---------------- host launch ----------------
extern "C" void kernel_launch(void* const* d_in, const int* in_sizes, int n_in,
                              void* d_out, int out_size, void* d_ws, size_t ws_size,
                              hipStream_t stream) {
  (void)in_sizes; (void)n_in; (void)out_size; (void)ws_size;
  const float* gene = (const float*)d_in[0];
  const float* sub  = (const float*)d_in[1];
  const float* Wq   = (const float*)d_in[2];
  const float* Wk   = (const float*)d_in[3];
  const float* Wv   = (const float*)d_in[4];
  const float* Wo   = (const float*)d_in[5];
  const float* lq1  = (const float*)d_in[6];
  const float* lk1  = (const float*)d_in[7];
  const float* lq2  = (const float*)d_in[8];
  const float* lk2  = (const float*)d_in[9];
  const float* rmsw = (const float*)d_in[10];

  const size_t MG = 8 * 4096;     // 32768 query rows
  const size_t MS = 8 * 1024;     // 8192 key rows
  float* outp = (float*)d_out;                       // [32768,512]
  float* diff = outp + MG * 512;                     // [32768,1024]

  char* ws = (char*)d_ws;
  float* lam  = (float*)ws;
  bf16* geneB = (bf16*)(ws + 256);
  bf16* subB  = geneB + MG * 512;
  bf16* Qb    = subB  + MS * 512;
  bf16* Kb    = Qb    + MG * 512;
  bf16* Vt    = Kb    + MS * 512;       // [b][e][s] = 8 x 512 x 1024
  bf16* WqB   = Vt    + MS * 512;
  bf16* WkB   = WqB   + 512 * 512;
  bf16* WvB   = WkB   + 512 * 512;
  bf16* WoB   = WvB   + 512 * 512;
  bf16* Yb    = WoB   + 512 * 512;
  bf16* Pb    = Yb    + MG * 512;       // [32768,1024] bf16
  float* Ypre = (float*)(ws + 256);     // aliases geneB+subB+Qb (dead by then)

  lambda_kernel<<<1, 256, 0, stream>>>(lq1, lk1, lq2, lk2, lam);
  cvt_kernel<<<(int)(MG * 512 / 1024), 256, 0, stream>>>(gene, geneB, (int)(MG * 512 / 4));
  cvt_kernel<<<(int)(MS * 512 / 1024), 256, 0, stream>>>(sub,  subB,  (int)(MS * 512 / 4));
  cvt_kernel<<<256, 256, 0, stream>>>(Wq, WqB, 512 * 512 / 4);
  cvt_kernel<<<256, 256, 0, stream>>>(Wk, WkB, 512 * 512 / 4);
  cvt_kernel<<<256, 256, 0, stream>>>(Wv, WvB, 512 * 512 / 4);
  cvt_kernel<<<256, 256, 0, stream>>>(Wo, WoB, 512 * 512 / 4);

  // Q = gene @ Wq^T * h^-0.5 ; K = sub @ Wk^T ; V^T = (sub @ Wv^T)^T per batch
  gemm_nt_bf16<<<dim3(8, MG / 128), 256, 0, stream>>>(geneB, WqB, Qb, nullptr, 512, 0.0625f, 0);
  gemm_nt_bf16<<<dim3(8, MS / 128), 256, 0, stream>>>(subB,  WkB, Kb, nullptr, 512, 1.0f, 0);
  gemm_nt_bf16<<<dim3(8, MS / 128), 256, 0, stream>>>(subB,  WvB, Vt, nullptr, 512, 1.0f, 2);

  scores_softmax<<<(int)(MG / 16), 256, 0, stream>>>(Qb, Kb, lam, diff, Pb);

  // Ypre[b] = P[b] @ V[b]  (pure NT GEMM per batch: B rows are Vt[b][e][:])
  for (int b = 0; b < 8; ++b)
    gemm_nt_bf16<<<dim3(8, 32), 256, 0, stream>>>(
        Pb + (size_t)b * 4096 * 1024, Vt + (size_t)b * 512 * 1024,
        nullptr, Ypre + (size_t)b * 4096 * 512, 1024, 1.0f, 1);

  rms_kernel<<<(int)MG, 256, 0, stream>>>(Ypre, rmsw, Yb);
  // out = Y @ Wo^T (f32 into d_out)
  gemm_nt_bf16<<<dim3(8, MG / 128), 256, 0, stream>>>(Yb, WoB, nullptr, outp, 512, 1.0f, 1);
}
